// OC_NOCA_RSSM_16492674416878
// MI455X (gfx1250) — matrix operand order repaired and checked by reading, not compiled, not run on hardware
//
#include <hip/hip_runtime.h>
#include <hip/hip_bf16.h>

// SLATE decoder, incremental KV-cache decode, single persistent workgroup.
// All GEMMs via v_wmma_f32_16x16x32_bf16. Weights pre-packed to bf16 in the
// exact WMMA B-operand fragment order; activations kept LDS-resident in the
// A-operand fragment order, so the inner loop is b128 loads + wmma only.

#define Lc   4
#define Dm   256
#define NHc  8
#define DH   32
#define FFc  1024
#define Vv   2048
#define GENc 16
#define Bc   8
#define NSc  7
#define TMAX 17

typedef __attribute__((ext_vector_type(16))) __bf16 v16bf;
typedef __attribute__((ext_vector_type(8)))  float  v8f;

// ---- persistent device scratch (fully rewritten every call) ----
__device__ float g_enc[Bc*NSc*Dm];
__device__ float g_Kc [Lc*Bc*NSc*Dm];
__device__ float g_Vc [Lc*Bc*NSc*Dm];
__device__ float g_Ks [Lc*Bc*TMAX*Dm];
__device__ float g_Vs [Lc*Bc*TMAX*Dm];
__device__ float g_logits[Bc*Vv];

// ---- packed bf16 weights, WMMA B-fragment order [ntile][kchunk][lane][16] ----
__device__ __align__(32) __bf16 p_Wq_s[Lc*Dm*Dm];
__device__ __align__(32) __bf16 p_Wk_s[Lc*Dm*Dm];
__device__ __align__(32) __bf16 p_Wv_s[Lc*Dm*Dm];
__device__ __align__(32) __bf16 p_Wo_s[Lc*Dm*Dm];
__device__ __align__(32) __bf16 p_Wq_c[Lc*Dm*Dm];
__device__ __align__(32) __bf16 p_Wo_c[Lc*Dm*Dm];
__device__ __align__(32) __bf16 p_W1 [Lc*Dm*FFc];
__device__ __align__(32) __bf16 p_W2 [Lc*FFc*Dm];
__device__ __align__(32) __bf16 p_Wout[Dm*Vv];

// ---- WMMA 16x16x32 bf16 operand maps (CDNA5 ISA 7.12.2) ----
// A fragment element i of lane (m,half): K = akmap(i,half)
// inverse: given kk (0..31): half=(kk>>3)&1, i=(kk&7)|((kk&16)>>1)
// B fragment element i of lane (n,half): K = half*8 + (i&7) + (i>>3)*16  (same as A)
__device__ __forceinline__ int swz(int row, int k) {      // A-operand scatter index
  int c = k >> 5, kk = k & 31;
  int half = (kk >> 3) & 1;
  int i = (kk & 7) | ((kk & 16) >> 1);
  return c * 512 + (row + 16 * half) * 16 + i;
}

// One 16x16 output tile: A (LDS, fragment order) x B (global bf16, fragment order)
template<int NKC>
__device__ __forceinline__ v8f tile_gemm(const __bf16* Asw, const __bf16* Bt, int lane) {
  v8f c = {0.f, 0.f, 0.f, 0.f, 0.f, 0.f, 0.f, 0.f};
  const __bf16* ap = Asw + lane * 16;
  const __bf16* bp = Bt  + lane * 16;
#pragma unroll
  for (int k = 0; k < NKC; ++k) {
    v16bf a = *(const v16bf*)(ap + k * 512);
    v16bf b = *(const v16bf*)(bp + k * 512);
    c = __builtin_amdgcn_wmma_f32_16x16x32_bf16(false, a, false, b, (short)0, c, false, false);
  }
  return c;
}

__device__ __forceinline__ float wave_sum(float v) {
#pragma unroll
  for (int o = 16; o > 0; o >>= 1) v += __shfl_xor(v, o, 32);
  return v;
}

__device__ __forceinline__ void lnorm(float (*xs)[Dm], __bf16* ha,
                                      const float* w, const float* b, bool wb,
                                      int wave, int lane) {
  if (wave < Bc) {
    int bb = wave;
    float s = 0.f;
#pragma unroll
    for (int i = 0; i < 8; ++i) s += xs[bb][lane + 32 * i];
    s = wave_sum(s);
    float mean = s * (1.f / Dm);
    float vs = 0.f;
#pragma unroll
    for (int i = 0; i < 8; ++i) { float d = xs[bb][lane + 32 * i] - mean; vs += d * d; }
    vs = wave_sum(vs);
    float r = rsqrtf(vs * (1.f / Dm) + 1e-5f);
#pragma unroll
    for (int i = 0; i < 8; ++i) {
      int k = lane + 32 * i;
      float hv = (xs[bb][k] - mean) * r * w[k] + b[k];
      ha[swz(bb, k)] = (__bf16)hv;
      if (wb) xs[bb][k] = hv;
    }
  }
}

// ---------- weight repack: f32 row-major -> bf16 WMMA B-fragment tiles ----------
__global__ void repack_kernel(const float* src, int which, int K, int N, int nMats) {
  __bf16* dst;
  switch (which) {
    case 0: dst = p_Wq_s; break;
    case 1: dst = p_Wk_s; break;
    case 2: dst = p_Wv_s; break;
    case 3: dst = p_Wo_s; break;
    case 4: dst = p_Wq_c; break;
    case 5: dst = p_Wo_c; break;
    case 6: dst = p_W1;  break;
    case 7: dst = p_W2;  break;
    default: dst = p_Wout; break;
  }
  int perMat = K * N;
  int numKc = K >> 5;
  int g = blockIdx.x * blockDim.x + threadIdx.x;
  if (g >= nMats * perMat) return;
  int mat = g / perMat, rem = g % perMat;
  int perTile = numKc * 512;
  int tn = rem / perTile;
  int r2 = rem % perTile;
  int c = r2 >> 9;
  int l32 = (r2 >> 4) & 31;
  int i = r2 & 15;
  int n = tn * 16 + (l32 & 15);
  int k = c * 32 + (l32 >> 4) * 8 + (i & 7) + (i >> 3) * 16;   // B map: K = half*8 + (i&7) + (i>>3)*16
  dst[g] = (__bf16)src[mat * perMat + k * N + n];
}

// ---------- precompute: enc = slots @ W_slot_proj ----------
__global__ void enc_kernel(const float* slots, const float* Wsp) {
  __shared__ float srow[Dm];
  int b = blockIdx.x, c = threadIdx.x;
  for (int s = 0; s < NSc; ++s) {
    __syncthreads();
    srow[c] = slots[(b * NSc + s) * Dm + c];
    __syncthreads();
    float acc = 0.f;
    for (int k = 0; k < Dm; ++k) acc += srow[k] * Wsp[k * Dm + c];
    g_enc[(b * NSc + s) * Dm + c] = acc;
  }
}

// ---------- precompute: cross-attn K/V per layer ----------
__global__ void kvc_kernel(const float* Wk_c, const float* Wv_c) {
  __shared__ float erow[Dm];
  int l = blockIdx.x >> 3, b = blockIdx.x & 7, c = threadIdx.x;
  const float* Wk = Wk_c + l * Dm * Dm;
  const float* Wv = Wv_c + l * Dm * Dm;
  for (int s = 0; s < NSc; ++s) {
    __syncthreads();
    erow[c] = g_enc[(b * NSc + s) * Dm + c];
    __syncthreads();
    float ak = 0.f, av = 0.f;
    for (int k = 0; k < Dm; ++k) { float e = erow[k]; ak += e * Wk[k * Dm + c]; av += e * Wv[k * Dm + c]; }
    g_Kc[((l * Bc + b) * NSc + s) * Dm + c] = ak;
    g_Vc[((l * Bc + b) * NSc + s) * Dm + c] = av;
  }
}

__global__ void zero_out(float* out, int n) {
  int i = blockIdx.x * blockDim.x + threadIdx.x;
  if (i < n) out[i] = 0.f;
}

// ---------- persistent decode: 1 block, 32 waves, 16 autoregressive steps ----------
__global__ __launch_bounds__(1024) void decode_kernel(
    const float* dict_emb, const float* pe,
    const float* ln_self_w, const float* ln_self_b,
    const float* ln_cross_w, const float* ln_cross_b,
    const float* ln_ffn_w, const float* ln_ffn_b,
    const float* b1, const float* b2,
    const float* ln_f_w, const float* ln_f_b,
    float* out) {
  __shared__ float xs[Bc][Dm];                       // residual, f32
  __shared__ __align__(32) __bf16 ha[8 * 512];       // activations, A-fragment order (K<=256)
  __shared__ float q2[Bc][Dm];                       // scaled queries, f32
  __shared__ __align__(32) __bf16 fa[16 * 512];      // FFN hidden half, A-fragment order (K<=512)
  __shared__ int tokS[Bc];

  const int tid = threadIdx.x, lane = tid & 31, wave = tid >> 5;
  const int m = lane & 15, half = lane >> 4;
  const float QS = 0.17677669529663687f;  // 1/sqrt(32)

  for (int i = tid; i < 8 * 512; i += 1024)  ha[i] = (__bf16)0.f;
  for (int i = tid; i < 16 * 512; i += 1024) fa[i] = (__bf16)0.f;
  if (tid < Bc) tokS[tid] = 0;  // BOS
  __syncthreads();

  for (int t = 0; t < GENc; ++t) {
    // embedding: x = dict_emb[tok] + pe[t]
    for (int i = tid; i < Bc * Dm; i += 1024) {
      int bb = i >> 8, k = i & 255;
      xs[bb][k] = dict_emb[tokS[bb] * Dm + k] + pe[t * Dm + k];
    }
    __syncthreads();

    for (int l = 0; l < Lc; ++l) {
      // ---- self-attn LN (layer 0 normalizes residual in place)
      lnorm(xs, ha, ln_self_w + l * Dm, ln_self_b + l * Dm, l == 0, wave, lane);
      __syncthreads();

      // ---- q/k/v projections: 48 WMMA tiles
      for (int tt = wave; tt < 48; tt += 32) {
        int mat = tt >> 4, tn = tt & 15;
        const __bf16* W = (mat == 0 ? p_Wq_s : (mat == 1 ? p_Wk_s : p_Wv_s))
                          + l * Dm * Dm + tn * 8 * 512;
        v8f c = tile_gemm<8>(ha, W, lane);
        if (half == 0) {
          int n0 = tn * 16;
#pragma unroll
          for (int r = 0; r < 8; ++r) {
            float v = c[r];
            if (mat == 0)      q2[r][n0 + m] = v * QS;
            else if (mat == 1) g_Ks[((l * Bc + r) * TMAX + t) * Dm + n0 + m] = v;
            else               g_Vs[((l * Bc + r) * TMAX + t) * Dm + n0 + m] = v;
          }
        }
      }
      __threadfence(); __syncthreads();

      // ---- self attention over cached positions 0..t (64 (b,head) pairs)
      for (int pp = 0; pp < 2; ++pp) {
        int idx = wave + 32 * pp, bb = idx >> 3, hd = idx & 7;
        const float* Kr = g_Ks + (l * Bc + bb) * TMAX * Dm + hd * DH;
        const float* Vr = g_Vs + (l * Bc + bb) * TMAX * Dm + hd * DH;
        float s = -1e30f;
        if (lane <= t) {
          float acc = 0.f;
#pragma unroll
          for (int d = 0; d < DH; ++d) acc += q2[bb][hd * DH + d] * Kr[lane * Dm + d];
          s = acc;
        }
        float mx = s;
#pragma unroll
        for (int o = 16; o > 0; o >>= 1) mx = fmaxf(mx, __shfl_xor(mx, o, 32));
        float e = (lane <= t) ? __expf(s - mx) : 0.f;
        float p = e / wave_sum(e);
        float o = 0.f;
        for (int j = 0; j <= t; ++j) o += __shfl(p, j, 32) * Vr[j * Dm + lane];
        ha[swz(bb, hd * DH + lane)] = (__bf16)o;   // chunk hd, fragment order
      }
      __syncthreads();

      // ---- output projection, accumulate into residual
      if (wave < 16) {
        v8f c = tile_gemm<8>(ha, p_Wo_s + l * Dm * Dm + wave * 8 * 512, lane);
        if (half == 0) {
          int n0 = wave * 16;
#pragma unroll
          for (int r = 0; r < 8; ++r) xs[r][n0 + m] += c[r];
        }
      }
      __syncthreads();

      // ---- cross attention
      lnorm(xs, ha, ln_cross_w + l * Dm, ln_cross_b + l * Dm, false, wave, lane);
      __syncthreads();
      if (wave < 16) {
        v8f c = tile_gemm<8>(ha, p_Wq_c + l * Dm * Dm + wave * 8 * 512, lane);
        if (half == 0) {
          int n0 = wave * 16;
#pragma unroll
          for (int r = 0; r < 8; ++r) q2[r][n0 + m] = c[r] * QS;
        }
      }
      __syncthreads();
      for (int pp = 0; pp < 2; ++pp) {
        int idx = wave + 32 * pp, bb = idx >> 3, hd = idx & 7;
        const float* Kr = g_Kc + (l * Bc + bb) * NSc * Dm + hd * DH;
        const float* Vr = g_Vc + (l * Bc + bb) * NSc * Dm + hd * DH;
        float s = -1e30f;
        if (lane < NSc) {
          float acc = 0.f;
#pragma unroll
          for (int d = 0; d < DH; ++d) acc += q2[bb][hd * DH + d] * Kr[lane * Dm + d];
          s = acc;
        }
        float mx = s;
#pragma unroll
        for (int o = 16; o > 0; o >>= 1) mx = fmaxf(mx, __shfl_xor(mx, o, 32));
        float e = (lane < NSc) ? __expf(s - mx) : 0.f;
        float p = e / wave_sum(e);
        float o = 0.f;
        for (int j = 0; j < NSc; ++j) o += __shfl(p, j, 32) * Vr[j * Dm + lane];
        ha[swz(bb, hd * DH + lane)] = (__bf16)o;
      }
      __syncthreads();
      if (wave < 16) {
        v8f c = tile_gemm<8>(ha, p_Wo_c + l * Dm * Dm + wave * 8 * 512, lane);
        if (half == 0) {
          int n0 = wave * 16;
#pragma unroll
          for (int r = 0; r < 8; ++r) xs[r][n0 + m] += c[r];
        }
      }
      __syncthreads();

      // ---- FFN (hidden dim split into 2 halves of 512)
      lnorm(xs, ha, ln_ffn_w + l * Dm, ln_ffn_b + l * Dm, false, wave, lane);
      __syncthreads();
      for (int nh = 0; nh < 2; ++nh) {
        { // FFN1: 32 tiles, K=256; output scattered to A-fragment order
          int tn = nh * 32 + wave;   // global n-tile in [0,64)
          v8f c = tile_gemm<8>(ha, p_W1 + l * Dm * FFc + tn * 8 * 512, lane);
          if (half == 0) {
            int n0 = wave * 16;
#pragma unroll
            for (int r = 0; r < 8; ++r) {
              float v = c[r] + b1[l * FFc + nh * 512 + n0 + m];
              fa[swz(r, n0 + m)] = (__bf16)fmaxf(v, 0.f);
            }
          }
        }
        __syncthreads();
        if (wave < 16) { // FFN2: 16 tiles, K=512 (chunks nh*16..nh*16+15)
          const __bf16* W = p_W2 + l * FFc * Dm + wave * 32 * 512 + nh * 16 * 512;
          v8f c = tile_gemm<16>(fa, W, lane);
          if (half == 0) {
            int n0 = wave * 16;
#pragma unroll
            for (int r = 0; r < 8; ++r)
              xs[r][n0 + m] += c[r] + (nh == 0 ? b2[l * Dm + n0 + m] : 0.f);
          }
        }
        __syncthreads();
      }
    } // layers

    // ---- final LN + logits (128 tiles over V=2048)
    lnorm(xs, ha, ln_f_w, ln_f_b, false, wave, lane);
    __syncthreads();
    for (int tn = wave; tn < 128; tn += 32) {
      v8f c = tile_gemm<8>(ha, p_Wout + tn * 8 * 512, lane);
      if (half == 0) {
        int n0 = tn * 16;
#pragma unroll
        for (int r = 0; r < 8; ++r) g_logits[r * Vv + n0 + m] = c[r];
      }
    }
    __threadfence(); __syncthreads();

    // ---- log_softmax + argmax, one wave per batch row
    if (wave < Bc) {
      int bb = wave;
      const float* lg = g_logits + bb * Vv;
      float mx = -1e30f; int mi = 0;
      for (int i = lane; i < Vv; i += 32) { float v = lg[i]; if (v > mx) { mx = v; mi = i; } }
#pragma unroll
      for (int o = 16; o > 0; o >>= 1) {
        float ov = __shfl_xor(mx, o, 32); int oi = __shfl_xor(mi, o, 32);
        if (ov > mx || (ov == mx && oi < mi)) { mx = ov; mi = oi; }
      }
      float se = 0.f;
      for (int i = lane; i < Vv; i += 32) se += __expf(lg[i] - mx);
      float lse = logf(wave_sum(se)) + mx;
      float* outLP = out + Bc * Vv * GENc + (bb * GENc + t) * Vv;
      for (int i = lane; i < Vv; i += 32) outLP[i] = lg[i] - lse;
      if (lane == 0) {
        tokS[bb] = mi + 1;                       // +1: BOS column prepended
        out[(bb * Vv + mi) * GENc + t] = 1.0f;   // z_img[b, v, he, we]
      }
    }
    __syncthreads();
  }
}

extern "C" void kernel_launch(void* const* d_in, const int* in_sizes, int n_in,
                              void* d_out, int out_size, void* d_ws, size_t ws_size,
                              hipStream_t stream) {
  (void)in_sizes; (void)n_in; (void)d_ws; (void)ws_size;
  const float* slots  = (const float*)d_in[0];
  const float* Wsp    = (const float*)d_in[1];
  const float* dict   = (const float*)d_in[2];
  const float* pe     = (const float*)d_in[3];
  const float* lnsw   = (const float*)d_in[4];
  const float* lnsb   = (const float*)d_in[5];
  const float* Wq_s   = (const float*)d_in[6];
  const float* Wk_s   = (const float*)d_in[7];
  const float* Wv_s   = (const float*)d_in[8];
  const float* Wo_s   = (const float*)d_in[9];
  const float* lncw   = (const float*)d_in[10];
  const float* lncb   = (const float*)d_in[11];
  const float* Wq_c   = (const float*)d_in[12];
  const float* Wk_c   = (const float*)d_in[13];
  const float* Wv_c   = (const float*)d_in[14];
  const float* Wo_c   = (const float*)d_in[15];
  const float* lnfw   = (const float*)d_in[16];
  const float* lnfb   = (const float*)d_in[17];
  const float* W1     = (const float*)d_in[18];
  const float* b1     = (const float*)d_in[19];
  const float* W2     = (const float*)d_in[20];
  const float* b2     = (const float*)d_in[21];
  const float* lnFw   = (const float*)d_in[22];
  const float* lnFb   = (const float*)d_in[23];
  const float* W_out  = (const float*)d_in[24];
  float* out = (float*)d_out;

  zero_out<<<(out_size + 1023) / 1024, 1024, 0, stream>>>(out, out_size);

  // weight repacks (one-time per call; L2-resident afterwards)
  repack_kernel<<<1024, 256, 0, stream>>>(Wq_s, 0, Dm, Dm, Lc);
  repack_kernel<<<1024, 256, 0, stream>>>(Wk_s, 1, Dm, Dm, Lc);
  repack_kernel<<<1024, 256, 0, stream>>>(Wv_s, 2, Dm, Dm, Lc);
  repack_kernel<<<1024, 256, 0, stream>>>(Wo_s, 3, Dm, Dm, Lc);
  repack_kernel<<<1024, 256, 0, stream>>>(Wq_c, 4, Dm, Dm, Lc);
  repack_kernel<<<1024, 256, 0, stream>>>(Wo_c, 5, Dm, Dm, Lc);
  repack_kernel<<<4096, 256, 0, stream>>>(W1, 6, Dm, FFc, Lc);
  repack_kernel<<<4096, 256, 0, stream>>>(W2, 7, FFc, Dm, Lc);
  repack_kernel<<<2048, 256, 0, stream>>>(W_out, 8, Dm, Vv, 1);

  enc_kernel<<<Bc, Dm, 0, stream>>>(slots, Wsp);
  kvc_kernel<<<Lc * Bc, Dm, 0, stream>>>(Wk_c, Wv_c);
  decode_kernel<<<1, 1024, 0, stream>>>(dict, pe, lnsw, lnsb,
                                        lncw, lncb, lnfw, lnfb,
                                        b1, b2, lnFw, lnFb, out);
}
